// MultimodalGraphModel_78383153152241
// MI455X (gfx1250) — matrix operand order repaired and checked by reading, not compiled, not run on hardware
//
#include <hip/hip_runtime.h>

// ---------------- types ----------------
typedef __bf16 bf16_t;
typedef __attribute__((ext_vector_type(16))) bf16_t v16bf;
typedef __attribute__((ext_vector_type(8)))  float  v8f;
typedef __attribute__((ext_vector_type(8)))  unsigned short us8;

union FragU { us8 u[2]; v16bf v; };
struct KFrags { FragU a0, a1, b0, b1, b2, b3; };

static __device__ __forceinline__ unsigned short f2bf(float f) {
    unsigned u = __float_as_uint(f);
    u += 0x7FFFu + ((u >> 16) & 1u);   // round-to-nearest-even
    return (unsigned short)(u >> 16);
}
static __device__ __forceinline__ unsigned f2mono(float f) {
    unsigned u = __float_as_uint(f);
    return (u & 0x80000000u) ? ~u : (u | 0x80000000u);
}
static __device__ __forceinline__ float mono2f(unsigned m) {
    return (m & 0x80000000u) ? __uint_as_float(m & 0x7FFFFFFFu) : __uint_as_float(~m);
}

// ---------------- utility kernels ----------------
__global__ void zero_f32(float* __restrict__ p, long long n) {
    long long i = (long long)blockIdx.x * blockDim.x + threadIdx.x;
    long long s = (long long)gridDim.x * blockDim.x;
    for (; i < n; i += s) p[i] = 0.0f;
}

__global__ void cvt_f32_bf16(const float* __restrict__ s, unsigned short* __restrict__ d, long long n) {
    long long i = (long long)blockIdx.x * blockDim.x + threadIdx.x;
    long long st = (long long)gridDim.x * blockDim.x;
    for (; i < n; i += st) d[i] = f2bf(s[i]);
}

// Repack row-major [K, ncols] f32 -> wave32 WMMA-B fragment layout (bf16).
// Tile (ko,no) is 512 contiguous halves; lane l holds B[k0+(l>>4)*16 + j, n0+(l&15)], j=0..15.
__global__ void pack_b_frag(const float* __restrict__ src, unsigned short* __restrict__ dst,
                            int K, int ncols, int ncolsTotal, int colOffset) {
    long long total = (long long)K * ncols;
    int nNtTot = ncolsTotal >> 4;
    long long i = (long long)blockIdx.x * blockDim.x + threadIdx.x;
    long long st = (long long)gridDim.x * blockDim.x;
    for (; i < total; i += st) {
        int k = (int)(i / ncols), n = (int)(i % ncols);
        int gn = n + colOffset;
        int ko = k >> 5, kin = k & 31, no = gn >> 4, nin = gn & 15;
        int lane = ((kin >> 4) << 4) | nin;
        int j = kin & 15;
        dst[(((long long)ko * nNtTot + no) << 9) + lane * 16 + j] = f2bf(src[i]);
    }
}

__global__ void bias_concat(const float* bq, const float* bk, const float* bv,
                            const float* bs, float* __restrict__ o) {
    int i = blockIdx.x * blockDim.x + threadIdx.x;
    if (i < 1024)
        o[i] = (i < 256) ? bq[i] : (i < 512) ? bk[i - 256] : (i < 768) ? bv[i - 512] : bs[i - 768];
}

// ---------------- WMMA GEMM helpers ----------------
static __device__ __forceinline__ void load_kstep(KFrags& f,
        const unsigned short* __restrict__ Ap0, const unsigned short* __restrict__ Ap1,
        const unsigned short* __restrict__ Bb, long long bstep, int ko) {
    const unsigned short* pa0 = Ap0 + ko * 32;
    const unsigned short* pa1 = Ap1 + ko * 32;
    const unsigned short* pb  = Bb + (long long)ko * bstep;
    f.a0.u[0] = *(const us8*)(pa0);        f.a0.u[1] = *(const us8*)(pa0 + 8);
    f.a1.u[0] = *(const us8*)(pa1);        f.a1.u[1] = *(const us8*)(pa1 + 8);
    f.b0.u[0] = *(const us8*)(pb);         f.b0.u[1] = *(const us8*)(pb + 8);
    f.b1.u[0] = *(const us8*)(pb + 512);   f.b1.u[1] = *(const us8*)(pb + 520);
    f.b2.u[0] = *(const us8*)(pb + 1024);  f.b2.u[1] = *(const us8*)(pb + 1032);
    f.b3.u[0] = *(const us8*)(pb + 1536);  f.b3.u[1] = *(const us8*)(pb + 1544);
}

static __device__ __forceinline__ void mma8(v8f acc0[4], v8f acc1[4], const KFrags& f) {
    acc0[0] = __builtin_amdgcn_wmma_f32_16x16x32_bf16(false, f.a0.v, false, f.b0.v, (short)0, acc0[0], false, false);
    acc1[0] = __builtin_amdgcn_wmma_f32_16x16x32_bf16(false, f.a1.v, false, f.b0.v, (short)0, acc1[0], false, false);
    acc0[1] = __builtin_amdgcn_wmma_f32_16x16x32_bf16(false, f.a0.v, false, f.b1.v, (short)0, acc0[1], false, false);
    acc1[1] = __builtin_amdgcn_wmma_f32_16x16x32_bf16(false, f.a1.v, false, f.b1.v, (short)0, acc1[1], false, false);
    acc0[2] = __builtin_amdgcn_wmma_f32_16x16x32_bf16(false, f.a0.v, false, f.b2.v, (short)0, acc0[2], false, false);
    acc1[2] = __builtin_amdgcn_wmma_f32_16x16x32_bf16(false, f.a1.v, false, f.b2.v, (short)0, acc1[2], false, false);
    acc0[3] = __builtin_amdgcn_wmma_f32_16x16x32_bf16(false, f.a0.v, false, f.b3.v, (short)0, acc0[3], false, false);
    acc1[3] = __builtin_amdgcn_wmma_f32_16x16x32_bf16(false, f.a1.v, false, f.b3.v, (short)0, acc1[3], false, false);
}

// ---------------- WMMA GEMM ----------------
// C[M,ncols] = A(bf16 row-major [M,K]) * Bfrag, f32 accum.
// block = 256 threads = 8 waves; wave -> 32 rows (2 M-tiles) x 64 cols (4 N-tiles).
// Double-buffered K pipeline with named (register-resident) buffers; K/32 must be even.
__global__ void __launch_bounds__(256) gemm_bf16_wmma(
                               const unsigned short* __restrict__ A,
                               const unsigned short* __restrict__ Bfrag,
                               float* __restrict__ C,
                               int M, int K, int ncols,
                               long long bfragStrideZ, long long cStrideZ,
                               const float* __restrict__ bias) {
    Bfrag += (long long)blockIdx.z * bfragStrideZ;
    C     += (long long)blockIdx.z * cStrideZ;
    int wave = threadIdx.x >> 5;
    int lane = threadIdx.x & 31;
    int mt0 = (blockIdx.x * 8 + wave) * 2;          // first of two M-tiles
    if (mt0 * 16 >= M) return;                      // wave-uniform
    int mt1 = mt0 + 1;
    bool has1 = (mt1 * 16) < M;                     // wave-uniform
    int nt0 = blockIdx.y * 4;
    int nNt = ncols >> 4;

    v8f acc0[4] = {};
    v8f acc1[4] = {};

    int akoff = (lane >> 4) * 16;
    int r0 = mt0 * 16 + (lane & 15);
    int r1 = (has1 ? mt1 * 16 : mt0 * 16) + (lane & 15);  // clamp to valid memory
    const unsigned short* Ap0 = A + (long long)r0 * K + akoff;
    const unsigned short* Ap1 = A + (long long)r1 * K + akoff;
    const unsigned short* Bb  = Bfrag + ((long long)nt0 << 9) + lane * 16;
    long long bstep = (long long)nNt << 9;
    int kSteps = K >> 5;   // even (24 or 4)

    KFrags fA, fB;
    load_kstep(fA, Ap0, Ap1, Bb, bstep, 0);
    load_kstep(fB, Ap0, Ap1, Bb, bstep, 1);

    for (int ko = 0; ko + 2 < kSteps; ko += 2) {
        mma8(acc0, acc1, fA);
        load_kstep(fA, Ap0, Ap1, Bb, bstep, ko + 2);
        mma8(acc0, acc1, fB);
        load_kstep(fB, Ap0, Ap1, Bb, bstep, ko + 3);
    }
    mma8(acc0, acc1, fA);
    mma8(acc0, acc1, fB);

    int ccol = lane & 15;
    int rb0 = mt0 * 16 + (lane >> 4) * 8;
    int rb1 = mt1 * 16 + (lane >> 4) * 8;
#pragma unroll
    for (int t = 0; t < 4; ++t) {
        int col = (nt0 + t) * 16 + ccol;
        float badd = bias ? bias[col] : 0.0f;
#pragma unroll
        for (int r = 0; r < 8; ++r)
            C[(long long)(rb0 + r) * ncols + col] = acc0[t][r] + badd;
        if (has1) {
#pragma unroll
            for (int r = 0; r < 8; ++r)
                C[(long long)(rb1 + r) * ncols + col] = acc1[t][r] + badd;
        }
    }
}

// ---------------- RGCN edge aggregation ----------------
// sums layout: [R][N][128]; xw layout: [R][N][128]
__global__ void rgcn_scatter(const int* __restrict__ src, const int* __restrict__ dst,
                             const int* __restrict__ etype,
                             const float* __restrict__ xw,
                             float* __restrict__ sums, float* __restrict__ cnt,
                             int E, int Nn) {
    long long total = (long long)E * 32;
    long long i = (long long)blockIdx.x * blockDim.x + threadIdx.x;
    long long st = (long long)gridDim.x * blockDim.x;
    for (; i < total; i += st) {
        int e = (int)(i >> 5);
        int c4 = (int)(i & 31) << 2;
        int s = src[e], d = dst[e], r = etype[e];
        const float4 m = *(const float4*)(xw + (((long long)r * Nn + s) << 7) + c4);
        float* o = sums + (((long long)r * Nn + d) << 7) + c4;
        atomicAdd(o + 0, m.x); atomicAdd(o + 1, m.y);
        atomicAdd(o + 2, m.z); atomicAdd(o + 3, m.w);
        if (c4 == 0) atomicAdd(cnt + (long long)r * Nn + d, 1.0f);
    }
}

__global__ void h_finalize(const float* __restrict__ hroot, const float* __restrict__ sums,
                           const float* __restrict__ cnt, const float* __restrict__ b_rgcn,
                           unsigned short* __restrict__ hbf, int Nn) {
    long long total = (long long)Nn * 128;
    long long i = (long long)blockIdx.x * blockDim.x + threadIdx.x;
    long long st = (long long)gridDim.x * blockDim.x;
    for (; i < total; i += st) {
        int n = (int)(i >> 7), c = (int)(i & 127);
        float acc = hroot[i] + b_rgcn[c];
#pragma unroll
        for (int r = 0; r < 6; ++r) {
            long long sidx = (long long)r * Nn + n;
            acc += sums[(sidx << 7) + c] / fmaxf(cnt[sidx], 1.0f);
        }
        hbf[i] = f2bf(fmaxf(acc, 0.0f));
    }
}

// ---------------- TransformerConv edge kernels ----------------
// qkvs layout per node: [q(256) | k(256) | v(256) | skip+agg(256)], row stride 1024.
__global__ void attn_alpha(const int* __restrict__ src, const int* __restrict__ dst,
                           const float* __restrict__ qkvs,
                           float* __restrict__ alpha, unsigned* __restrict__ amax, int E) {
    long long total = (long long)E * 4;
    long long i = (long long)blockIdx.x * blockDim.x + threadIdx.x;
    long long st = (long long)gridDim.x * blockDim.x;
    for (; i < total; i += st) {
        int e = (int)(i >> 2), hh = (int)(i & 3);
        int s = src[e], d = dst[e];
        const float4* qp = (const float4*)(qkvs + ((long long)d << 10) + hh * 64);
        const float4* kp = (const float4*)(qkvs + ((long long)s << 10) + 256 + hh * 64);
        float acc = 0.0f;
#pragma unroll
        for (int j = 0; j < 16; ++j) {
            float4 q = qp[j], k = kp[j];
            acc += q.x * k.x + q.y * k.y + q.z * k.z + q.w * k.w;
        }
        acc *= 0.125f;  // 1/sqrt(64)
        alpha[i] = acc;
        atomicMax(&amax[(long long)d * 4 + hh], f2mono(acc));
    }
}

__global__ void attn_exp(const int* __restrict__ dst, float* __restrict__ alpha,
                         const unsigned* __restrict__ amax, float* __restrict__ denom, int E) {
    long long total = (long long)E * 4;
    long long i = (long long)blockIdx.x * blockDim.x + threadIdx.x;
    long long st = (long long)gridDim.x * blockDim.x;
    for (; i < total; i += st) {
        int e = (int)(i >> 2), hh = (int)(i & 3);
        int d = dst[e];
        float ex = expf(alpha[i] - mono2f(amax[(long long)d * 4 + hh]));
        alpha[i] = ex;
        atomicAdd(&denom[(long long)d * 4 + hh], ex);
    }
}

__global__ void attn_agg(const int* __restrict__ src, const int* __restrict__ dst,
                         const float* __restrict__ ex, const float* __restrict__ denom,
                         float* __restrict__ qkvs, int E) {
    long long total = (long long)E * 64;
    long long i = (long long)blockIdx.x * blockDim.x + threadIdx.x;
    long long st = (long long)gridDim.x * blockDim.x;
    for (; i < total; i += st) {
        int e = (int)(i >> 6);
        int t = (int)(i & 63);
        int hh = t >> 4, c4 = (t & 15) << 2;
        int s = src[e], d = dst[e];
        float attn = ex[(long long)e * 4 + hh] /
                     fmaxf(denom[(long long)d * 4 + hh], 1e-16f);
        const float4 v = *(const float4*)(qkvs + ((long long)s << 10) + 512 + hh * 64 + c4);
        float* o = qkvs + ((long long)d << 10) + 768 + hh * 64 + c4;
        atomicAdd(o + 0, attn * v.x); atomicAdd(o + 1, attn * v.y);
        atomicAdd(o + 2, attn * v.z); atomicAdd(o + 3, attn * v.w);
    }
}

__global__ void relu_mean_pool(const float* __restrict__ qkvs, float* __restrict__ pooled, int Nn) {
    int c = threadIdx.x;  // 256 threads
    float acc = 0.0f;
    for (int n = blockIdx.x; n < Nn; n += gridDim.x)
        acc += fmaxf(qkvs[((long long)n << 10) + 768 + c], 0.0f);
    atomicAdd(&pooled[c], acc);
}

__global__ void final_fc(const float* __restrict__ pooled, const float* __restrict__ Wfc,
                         const float* __restrict__ bfc, float* __restrict__ out, int Nn) {
    int j = threadIdx.x;
    if (j < 3) {
        float acc = bfc[j];
        float inv = 1.0f / (float)Nn;
        for (int c = 0; c < 256; ++c) acc += pooled[c] * inv * Wfc[c * 3 + j];
        out[j] = acc;
    }
}

// ---------------- host ----------------
extern "C" void kernel_launch(void* const* d_in, const int* in_sizes, int n_in,
                              void* d_out, int out_size, void* d_ws, size_t ws_size,
                              hipStream_t stream) {
    const float* x      = (const float*)d_in[0];
    const int*   eidx   = (const int*)d_in[1];
    const int*   etype  = (const int*)d_in[2];
    const float* W_rel  = (const float*)d_in[3];
    const float* W_root = (const float*)d_in[4];
    const float* b_rgcn = (const float*)d_in[5];
    const float* Wq = (const float*)d_in[6];  const float* bq = (const float*)d_in[7];
    const float* Wk = (const float*)d_in[8];  const float* bk = (const float*)d_in[9];
    const float* Wv = (const float*)d_in[10]; const float* bv = (const float*)d_in[11];
    const float* Ws = (const float*)d_in[12]; const float* bs = (const float*)d_in[13];
    const float* Wfc = (const float*)d_in[14]; const float* bfc = (const float*)d_in[15];
    float* out = (float*)d_out;

    const int Nn = 50000, E = 600000, DIN = 768, DH = 128, R = 6;

    char* ws = (char*)d_ws;
    size_t off = 0;
    auto alloc = [&](size_t bytes) -> void* {
        void* p = ws + off;
        off = (off + bytes + 255) & ~(size_t)255;
        return p;
    };

    // xwAll: planes 0..5 = x@W_rel[r], plane 6 = x@W_root  (single fused GEMM)
    float* xwAll  = (float*)alloc((size_t)(R + 1) * Nn * DH * 4);
    float* hroot  = xwAll + (size_t)R * Nn * DH;
    float* sums   = (float*)alloc((size_t)R * Nn * DH * 4);   // adjacent to cnt
    float* cnt    = (float*)alloc((size_t)R * Nn * 4);
    float* qkvs   = (float*)alloc((size_t)Nn * 1024 * 4);     // [N][q|k|v|skip]
    float* alpha  = (float*)alloc((size_t)E * 4 * 4);         // [E][H] -> reused as ex
    unsigned* amaxu = (unsigned*)alloc((size_t)Nn * 4 * 4);   // monotonic-mapped max
    float* denom  = (float*)alloc((size_t)Nn * 4 * 4);
    float* pooled = (float*)alloc(256 * 4);
    float* biascat = (float*)alloc(1024 * 4);
    unsigned short* xbf      = (unsigned short*)alloc((size_t)Nn * DIN * 2);
    unsigned short* hbf      = (unsigned short*)alloc((size_t)Nn * DH * 2);
    unsigned short* fragRR   = (unsigned short*)alloc((size_t)(R + 1) * DIN * DH * 2); // rel + root
    unsigned short* fragQKVS = (unsigned short*)alloc((size_t)DH * 1024 * 2);
    (void)ws_size; (void)n_in; (void)out_size; (void)in_sizes;

    // zero accumulators (sums+cnt contiguous; amaxu+denom+pooled contiguous)
    zero_f32<<<4096, 256, 0, stream>>>(sums, (long long)R * Nn * DH + (long long)R * Nn);
    zero_f32<<<256, 256, 0, stream>>>((float*)amaxu, (long long)Nn * 4 * 2 + 256);

    // precision prep: bf16 activations + fragment-packed bf16 weights
    cvt_f32_bf16<<<8192, 256, 0, stream>>>(x, xbf, (long long)Nn * DIN);
    for (int r = 0; r < R; ++r)
        pack_b_frag<<<512, 256, 0, stream>>>(W_rel + (size_t)r * DIN * DH,
                                             fragRR + (size_t)r * DIN * DH,
                                             DIN, DH, DH, 0);
    pack_b_frag<<<512, 256, 0, stream>>>(W_root, fragRR + (size_t)R * DIN * DH, DIN, DH, DH, 0);
    pack_b_frag<<<256, 256, 0, stream>>>(Wq, fragQKVS, DH, 256, 1024, 0);
    pack_b_frag<<<256, 256, 0, stream>>>(Wk, fragQKVS, DH, 256, 1024, 256);
    pack_b_frag<<<256, 256, 0, stream>>>(Wv, fragQKVS, DH, 256, 1024, 512);
    pack_b_frag<<<256, 256, 0, stream>>>(Ws, fragQKVS, DH, 256, 1024, 768);
    bias_concat<<<4, 256, 0, stream>>>(bq, bk, bv, bs, biascat);

    int mtiles = (Nn + 15) / 16;                 // 3125
    int mblocks = (mtiles + 15) / 16;            // 8 waves x 2 tiles per block

    // [xw planes | hroot] = x @ [W_rel[0..5] | W_root]   (WMMA bf16, z = plane)
    dim3 g1(mblocks, DH / 64, R + 1);
    gemm_bf16_wmma<<<g1, 256, 0, stream>>>(xbf, fragRR, xwAll, Nn, DIN, DH,
                                           (long long)DIN * DH, (long long)Nn * DH, nullptr);

    // edge message scatter (mean numerator + counts)
    rgcn_scatter<<<8192, 256, 0, stream>>>(eidx, eidx + E, etype, xwAll, sums, cnt, E, Nn);

    // h = relu(mean-agg + hroot + b); emit bf16 for next GEMM
    h_finalize<<<4096, 256, 0, stream>>>(hroot, sums, cnt, b_rgcn, hbf, Nn);

    // [q|k|v|skip] = h @ [Wq|Wk|Wv|Wskip] + bias   (WMMA bf16)
    dim3 g3(mblocks, 1024 / 64, 1);
    gemm_bf16_wmma<<<g3, 256, 0, stream>>>(hbf, fragQKVS, qkvs, Nn, DH, 1024, 0, 0, biascat);

    // per-edge multi-head attention
    attn_alpha<<<4096, 256, 0, stream>>>(eidx, eidx + E, qkvs, alpha, amaxu, E);
    attn_exp<<<4096, 256, 0, stream>>>(eidx + E, alpha, amaxu, denom, E);
    attn_agg<<<8192, 256, 0, stream>>>(eidx, eidx + E, alpha, denom, qkvs, E);

    // relu + mean pool + fc
    relu_mean_pool<<<256, 256, 0, stream>>>(qkvs, pooled, Nn);
    final_fc<<<1, 32, 0, stream>>>(pooled, Wfc, bfc, out, Nn);
}
